// FinalCoarseToFineSemanticUpModule_84464826843864
// MI455X (gfx1250) — matrix operand order repaired and checked by reading, not compiled, not run on hardware
//
#include <hip/hip_runtime.h>
#include <hip/hip_bf16.h>
#include <math.h>

// ---------------- shapes (fixed by the reference harness) ----------------
#define BATCH   8
#define NFINE   8192        // N (2^13)
#define KCOARSE 1024        // K
#define CDIM    512         // C
#define HALFD   256
#define NEGBIG  1.0e9f
#define SCALE_QK 0.044194173824159216f   // 1/sqrt(512)
#define TWO_PI_F 6.283185307179586f

typedef __attribute__((ext_vector_type(16))) __bf16 v16bf;
typedef __attribute__((ext_vector_type(8)))  float  v8f;
typedef __attribute__((ext_vector_type(4)))  unsigned int u32x4;
typedef __attribute__((ext_vector_type(8)))  int i32x8;
typedef __attribute__((ext_vector_type(4)))  int i32x4;

struct FragBits { uint4 lo, hi; };

// A-matrix 16x32 bf16 fragment (CDNA5 layout):
// lane L: m = L&15 ; lanes 0-15 hold K 0..7 & 16..23, lanes 16-31 hold K 8..15 & 24..31
static __device__ inline v16bf load_frag_A(const __bf16* tile_base, int ld, int kbase, int lane) {
    const int m    = lane & 15;
    const int koff = (lane >> 4) * 8;
    const __bf16* p = tile_base + (long)m * ld + kbase + koff;
    FragBits f;
    f.lo = *reinterpret_cast<const uint4*>(p);        // K = koff .. koff+7
    f.hi = *reinterpret_cast<const uint4*>(p + 16);   // K = koff+16 .. koff+23
    return __builtin_bit_cast(v16bf, f);
}

// B-matrix 32x16 bf16 fragment: lane L: n = L&15 ; lanes 0-15 hold K 0..7 & 16..23, lanes 16-31 K 8..15 & 24..31
static __device__ inline v16bf load_frag_B(const __bf16* tile_base, int ld, int kbase, int lane) {
    const int n    = lane & 15;
    const int koff = (lane >> 4) * 8;
    const __bf16* p = tile_base + (long)n * ld + kbase + koff;
    FragBits f;
    f.lo = *reinterpret_cast<const uint4*>(p);
    f.hi = *reinterpret_cast<const uint4*>(p + 16);
    return __builtin_bit_cast(v16bf, f);
}

static __device__ inline v8f wmma_bf16(v16bf a, v16bf b, v8f c) {
    return __builtin_amdgcn_wmma_f32_16x16x32_bf16(false, a, false, b, (short)0, c, false, false);
}

// ---- Tensor Data Mover: DMA a 2D bf16 tile (ROWS x W_ELEMS, row stride STRIDE_ELEMS)
// from global memory into LDS. D# per CDNA5 ISA ch.8 (group0/group1; groups 2/3 unused).
template <int W_ELEMS, int ROWS, int STRIDE_ELEMS>
static __device__ inline void tdm_load_2d(const __bf16* gsrc, unsigned lds_off) {
    unsigned long long ga = (unsigned long long)(size_t)gsrc;
    u32x4 g0 = { 1u,                                   // count=1 valid descriptor
                 lds_off,                              // lds_addr (bytes)
                 (unsigned)ga,                         // global_addr[31:0]
                 (((unsigned)(ga >> 32)) & 0x01FFFFFFu) | 0x80000000u }; // addr[56:32] | type=2
    i32x8 g1 = {
        (int)(1u << 16),                                              // data_size = 2B (code 1)
        (int)(((unsigned)STRIDE_ELEMS & 0xFFFFu) << 16),              // tensor_dim0[15:0]
        (int)((((unsigned)STRIDE_ELEMS >> 16) & 0xFFFFu)
              | (((unsigned)ROWS & 0xFFFFu) << 16)),                  // dim0[31:16] | dim1[15:0]
        (int)((((unsigned)ROWS >> 16) & 0xFFFFu)
              | (((unsigned)W_ELEMS & 0xFFFFu) << 16)),               // dim1[31:16] | tile_dim0
        ROWS,                                                         // tile_dim1 (tile_dim2=0)
        STRIDE_ELEMS,                                                 // tensor_dim0_stride[31:0]
        (int)(((unsigned)STRIDE_ELEMS & 0xFFFFu) << 16),              // dim0_stride[47:32]=0 | dim1_stride[15:0]
        0 };
    i32x4 z4 = { 0, 0, 0, 0 };
#if __clang_major__ >= 23
    i32x8 z8 = { 0, 0, 0, 0, 0, 0, 0, 0 };
    __builtin_amdgcn_tensor_load_to_lds(g0, g1, z4, z4, z8, 0);
#else
    __builtin_amdgcn_tensor_load_to_lds(g0, g1, z4, z4, 0);
#endif
}

// ---------------- denom[b] = max(max(sum(node_mask),1)-1, 1) ----------------
__global__ __launch_bounds__(256) void reduce_mask_kernel(const float* __restrict__ node_mask,
                                                          float* __restrict__ denomArr) {
    __shared__ float red[256];
    const int b = blockIdx.x, t = threadIdx.x;
    float s = 0.0f;
    for (int i = t; i < NFINE; i += 256) s += node_mask[(long)b * NFINE + i];
    red[t] = s; __syncthreads();
    for (int off = 128; off > 0; off >>= 1) {
        if (t < off) red[t] += red[t + off];
        __syncthreads();
    }
    if (t == 0) {
        float L = fmaxf(red[0], 1.0f);
        denomArr[b] = fmaxf(L - 1.0f, 1.0f);
    }
}

// ---------------- elementwise f32 -> bf16 ----------------
__global__ void f32_to_bf16_kernel(const float* __restrict__ in, __bf16* __restrict__ out, long n) {
    long i = (long)blockIdx.x * blockDim.x + threadIdx.x;
    const long stride = (long)gridDim.x * blockDim.x;
    for (; i < n; i += stride) out[i] = (__bf16)in[i];
}

// ---------------- Fourier embedding + LayerNorm -> q0 (bf16) ----------------
__global__ __launch_bounds__(256) void fourier_ln_kernel(
    const int* __restrict__ res_idx, const float* __restrict__ Wff,
    const float* __restrict__ node_mask, const float* __restrict__ denomArr,
    const float* __restrict__ g1, const float* __restrict__ b1,
    __bf16* __restrict__ q0) {
    __shared__ float s_sum[256];
    __shared__ float s_sq[256];
    const int t = threadIdx.x;
    const long row = blockIdx.x;            // b*NFINE + n
    const int b = blockIdx.x >> 13;
    const float denom = denomArr[b];
    float pos = (float)res_idx[row] / denom;
    pos = fminf(fmaxf(pos, 0.0f), 1.0f);
    const float m = node_mask[row];
    const float ang = TWO_PI_F * pos * Wff[t];
    float sn, cs;
    sincosf(ang, &sn, &cs);
    cs *= m; sn *= m;
    s_sum[t] = cs + sn;
    s_sq[t]  = cs * cs + sn * sn;
    __syncthreads();
    for (int off = 128; off > 0; off >>= 1) {
        if (t < off) { s_sum[t] += s_sum[t + off]; s_sq[t] += s_sq[t + off]; }
        __syncthreads();
    }
    const float mean = s_sum[0] * (1.0f / 512.0f);
    const float var  = s_sq[0]  * (1.0f / 512.0f) - mean * mean;
    const float rstd = rsqrtf(var + 1e-5f);
    __bf16* o = q0 + row * CDIM;
    o[t]         = (__bf16)((cs - mean) * rstd * g1[t]         + b1[t]);
    o[t + HALFD] = (__bf16)((sn - mean) * rstd * g1[t + HALFD] + b1[t + HALFD]);
}

// ---------------- generic Y = X @ W^T via WMMA bf16, TDM-fed ----------------
// One wave per 16x128 output strip. Per k-chunk, the 8 B-tiles form one contiguous
// 128-row x 32-elem 2D region of W (row stride 512): DMA'd by the Tensor Data Mover
// into double-buffered LDS (2 x 8KB) with chunk-granular prefetch (TENSORcnt).
// A fragments (2 x b128 per chunk) stay register-pipelined from global.
__global__ __launch_bounds__(32) void gemm_xWT_bf16(
    const __bf16* __restrict__ A, const __bf16* __restrict__ W,
    __bf16* __restrict__ Ybf, float* __restrict__ Yf) {
    __shared__ __align__(16) __bf16 wbuf[2][128 * 32];   // 2 x 8KB W chunks
    const int lane  = threadIdx.x;
    const int strip = blockIdx.x & 3;
    const int mt    = blockIdx.x >> 2;
    const int row0  = mt * 16;
    const int col0  = strip * 128;
    const __bf16* Atile  = A + (long)row0 * CDIM;
    const __bf16* Wstrip = W + (long)col0 * CDIM;
    const unsigned lds_base = (unsigned)(size_t)(&wbuf[0][0]);

    // prologue: DMA W chunk 0 (rows col0..col0+127, k 0..31)
    tdm_load_2d<32, 128, CDIM>(Wstrip, lds_base);

    v8f acc[8] = {};
    v16bf a_cur = load_frag_A(Atile, CDIM, 0, lane);

    for (int kc = 0; kc < 16; ++kc) {
        const int cur = kc & 1;
        if (kc + 1 < 16) {
            asm volatile("s_wait_dscnt 0x0" ::: "memory"); // done reading buffer being refilled
            tdm_load_2d<32, 128, CDIM>(Wstrip + (kc + 1) * 32,
                                       lds_base + (unsigned)((cur ^ 1) * 128 * 32 * 2));
            asm volatile("s_wait_tensorcnt 0x1" ::: "memory");  // chunk kc has landed
        } else {
            asm volatile("s_wait_tensorcnt 0x0" ::: "memory");
        }
        v16bf a_nxt;
        if (kc + 1 < 16) a_nxt = load_frag_A(Atile, CDIM, (kc + 1) * 32, lane);
        const __bf16* wch = &wbuf[cur][0];
        #pragma unroll
        for (int t = 0; t < 8; ++t) {
            v16bf bm = load_frag_B(wch + t * 16 * 32, 32, 0, lane);  // ds_load_b128 x2
            acc[t] = wmma_bf16(a_cur, bm, acc[t]);
        }
        if (kc + 1 < 16) a_cur = a_nxt;
    }

    const int n  = lane & 15;
    const int mb = (lane >> 4) * 8;
    #pragma unroll
    for (int t = 0; t < 8; ++t) {
        #pragma unroll
        for (int r = 0; r < 8; ++r) {
            const float val = acc[t][r];
            const long idx = (long)(row0 + mb + r) * CDIM + (col0 + t * 16 + n);
            if (Ybf) Ybf[idx] = (__bf16)val;
            if (Yf)  Yf[idx]  = val;
        }
    }
}

// ---------------- fused logits (q @ k^T / sqrt(C)) + top-2 over K ----------------
// One wave per 16 fine rows. q fragments live in 128 VGPRs. k tiles (16x512 bf16 = 16KB)
// are DMA'd by the Tensor Data Mover into a double-buffered LDS region (TENSORcnt),
// giving a full-tile prefetch distance; WMMA B-fragments come from LDS (ds_load_b128).
__global__ __launch_bounds__(32) void logits_top2_kernel(
    const __bf16* __restrict__ Q, const __bf16* __restrict__ Km,
    const float* __restrict__ mask_parent, const float* __restrict__ node_mask,
    float* __restrict__ Bl, int* __restrict__ Pi) {
    __shared__ __align__(16) __bf16 kbuf[2][16 * CDIM];   // 2 x 16KB
    __shared__ float tile[256];
    const int lane = threadIdx.x;
    const int nt = blockIdx.x & ((NFINE / 16) - 1);
    const int b  = blockIdx.x >> 9;
    const __bf16* qtile = Q + (long)(b * NFINE + nt * 16) * CDIM;
    const __bf16* kb    = Km + (long)b * KCOARSE * CDIM;
    const float*  mp    = mask_parent + (long)b * KCOARSE;

    const unsigned lds_base = (unsigned)(size_t)(&kbuf[0][0]);

    v16bf afr[16];
    #pragma unroll
    for (int kc = 0; kc < 16; ++kc) afr[kc] = load_frag_A(qtile, CDIM, kc * 32, lane);

    float v1 = -3.0e38f, v2 = -3.0e38f;
    int   i1 = 0, i2 = 0;
    const int nloc = lane & 15;
    const int mb   = (lane >> 4) * 8;

    // prologue: DMA tile 0 into buffer 0
    tdm_load_2d<CDIM, 16, CDIM>(kb, lds_base);

    for (int kt = 0; kt < KCOARSE / 16; ++kt) {
        const int cur = kt & 1;
        if (kt + 1 < KCOARSE / 16) {
            asm volatile("s_wait_dscnt 0x0" ::: "memory");
            tdm_load_2d<CDIM, 16, CDIM>(kb + (long)((kt + 1) * 16) * CDIM,
                                        lds_base + (unsigned)((cur ^ 1) * 16 * CDIM * 2));
            asm volatile("s_wait_tensorcnt 0x1" ::: "memory");  // tile kt has landed
        } else {
            asm volatile("s_wait_tensorcnt 0x0" ::: "memory");
        }

        const __bf16* ktile = &kbuf[cur][0];
        v8f acc0 = {}, acc1 = {};
        #pragma unroll
        for (int kc = 0; kc < 16; ++kc) {
            v16bf bm = load_frag_B(ktile, CDIM, kc * 32, lane);   // ds_load_b128 x2
            if (kc & 1) acc1 = wmma_bf16(afr[kc], bm, acc1);
            else        acc0 = wmma_bf16(afr[kc], bm, acc0);
        }
        #pragma unroll
        for (int r = 0; r < 8; ++r) tile[(mb + r) * 16 + nloc] = acc0[r] + acc1[r];
        __syncthreads();
        if (lane < 16) {
            for (int j = 0; j < 16; ++j) {
                const int col = kt * 16 + j;
                const float val = tile[lane * 16 + j] * SCALE_QK + (mp[col] - 1.0f) * NEGBIG;
                if (val > v1)      { v2 = v1; i2 = i1; v1 = val; i1 = col; }
                else if (val > v2) { v2 = val; i2 = col; }
            }
        }
        __syncthreads();
    }
    if (lane < 16) {
        const int ng = nt * 16 + lane;
        const long row = (long)b * NFINE + ng;
        const float m = node_mask[row];
        const float e2 = __expf(v2 - v1);
        const float p1 = 1.0f / (1.0f + e2);
        const float p2 = e2 * p1;
        Bl[row * 2 + 0] = p1 * m;
        Bl[row * 2 + 1] = p2 * m;
        Pi[row * 2 + 0] = i1;
        Pi[row * 2 + 1] = i2;
    }
}

// ---------------- s0 = (B1*v[i1] + B2*v[i2]) * m ; s_fine = LN(s0)*g2+b2 ----------------
__global__ __launch_bounds__(256) void combine_ln_kernel(
    const float* __restrict__ V, const float* __restrict__ Bl, const int* __restrict__ Pi,
    const float* __restrict__ node_mask, const float* __restrict__ g2, const float* __restrict__ b2,
    float* __restrict__ out) {
    __shared__ float s_sum[256];
    __shared__ float s_sq[256];
    const int t = threadIdx.x;
    const long row = blockIdx.x;
    const int b = blockIdx.x >> 13;
    const float p1 = Bl[row * 2], p2 = Bl[row * 2 + 1];
    const int i1 = Pi[row * 2], i2 = Pi[row * 2 + 1];
    const float m = node_mask[row];
    const float* v1p = V + ((long)b * KCOARSE + i1) * CDIM;
    const float* v2p = V + ((long)b * KCOARSE + i2) * CDIM;
    const float x0 = (p1 * v1p[t]       + p2 * v2p[t])       * m;
    const float x1 = (p1 * v1p[t + 256] + p2 * v2p[t + 256]) * m;
    s_sum[t] = x0 + x1;
    s_sq[t]  = x0 * x0 + x1 * x1;
    __syncthreads();
    for (int off = 128; off > 0; off >>= 1) {
        if (t < off) { s_sum[t] += s_sum[t + off]; s_sq[t] += s_sq[t + off]; }
        __syncthreads();
    }
    const float mean = s_sum[0] * (1.0f / 512.0f);
    const float var  = s_sq[0]  * (1.0f / 512.0f) - mean * mean;
    const float rstd = rsqrtf(var + 1e-5f);
    float* o = out + row * CDIM;
    o[t]       = (x0 - mean) * rstd * g2[t]       + b2[t];
    o[t + 256] = (x1 - mean) * rstd * g2[t + 256] + b2[t + 256];
}

// ---------------- host launcher ----------------
extern "C" void kernel_launch(void* const* d_in, const int* in_sizes, int n_in,
                              void* d_out, int out_size, void* d_ws, size_t ws_size,
                              hipStream_t stream) {
    const float* s_parent    = (const float*)d_in[0];
    const float* Wff         = (const float*)d_in[1];
    const float* Wq          = (const float*)d_in[2];
    const float* Wk          = (const float*)d_in[3];
    const float* Wv          = (const float*)d_in[4];
    const float* g1          = (const float*)d_in[5];
    const float* b1          = (const float*)d_in[6];
    const float* g2          = (const float*)d_in[7];
    const float* b2          = (const float*)d_in[8];
    const float* mask_parent = (const float*)d_in[9];
    const float* node_mask   = (const float*)d_in[10];
    const int*   res_idx     = (const int*)d_in[11];

    char* ws = (char*)d_ws;
    size_t off = 0;
    auto carve = [&](size_t bytes) {
        void* p = ws + off;
        off = (off + bytes + 255) & ~(size_t)255;
        return p;
    };
    float*  denomArr = (float*) carve((size_t)BATCH * sizeof(float));
    __bf16* sp_bf    = (__bf16*)carve((size_t)BATCH * KCOARSE * CDIM * 2);
    __bf16* Wq_bf    = (__bf16*)carve((size_t)CDIM * CDIM * 2);
    __bf16* Wk_bf    = (__bf16*)carve((size_t)CDIM * CDIM * 2);
    __bf16* Wv_bf    = (__bf16*)carve((size_t)CDIM * CDIM * 2);
    __bf16* q0_bf    = (__bf16*)carve((size_t)BATCH * NFINE * CDIM * 2);
    __bf16* q_bf     = (__bf16*)carve((size_t)BATCH * NFINE * CDIM * 2);
    __bf16* k_bf     = (__bf16*)carve((size_t)BATCH * KCOARSE * CDIM * 2);
    float*  v_f      = (float*) carve((size_t)BATCH * KCOARSE * CDIM * 4);

    // output: s_fine [B*N*C] f32 ++ B_local [B*N*2] f32 ++ parent_idx [B*N*2] i32
    float* out_sfine = (float*)d_out;
    float* out_B     = out_sfine + (size_t)BATCH * NFINE * CDIM;
    int*   out_I     = (int*)(out_B + (size_t)BATCH * NFINE * 2);

    reduce_mask_kernel<<<BATCH, 256, 0, stream>>>(node_mask, denomArr);
    f32_to_bf16_kernel<<<2048, 256, 0, stream>>>(s_parent, sp_bf, (long)BATCH * KCOARSE * CDIM);
    f32_to_bf16_kernel<<<256, 256, 0, stream>>>(Wq, Wq_bf, (long)CDIM * CDIM);
    f32_to_bf16_kernel<<<256, 256, 0, stream>>>(Wk, Wk_bf, (long)CDIM * CDIM);
    f32_to_bf16_kernel<<<256, 256, 0, stream>>>(Wv, Wv_bf, (long)CDIM * CDIM);

    fourier_ln_kernel<<<BATCH * NFINE, 256, 0, stream>>>(res_idx, Wff, node_mask, denomArr,
                                                         g1, b1, q0_bf);

    gemm_xWT_bf16<<<(BATCH * NFINE   / 16) * 4, 32, 0, stream>>>(q0_bf, Wq_bf, q_bf, nullptr);
    gemm_xWT_bf16<<<(BATCH * KCOARSE / 16) * 4, 32, 0, stream>>>(sp_bf, Wk_bf, k_bf, nullptr);
    gemm_xWT_bf16<<<(BATCH * KCOARSE / 16) * 4, 32, 0, stream>>>(sp_bf, Wv_bf, nullptr, v_f);

    logits_top2_kernel<<<BATCH * (NFINE / 16), 32, 0, stream>>>(q_bf, k_bf, mask_parent,
                                                                node_mask, out_B, out_I);

    combine_ln_kernel<<<BATCH * NFINE, 256, 0, stream>>>(v_f, out_B, out_I, node_mask,
                                                         g2, b2, out_sfine);
}
